// Chemical_feature_interaction_36721970381123
// MI455X (gfx1250) — matrix operand order repaired and checked by reading, not compiled, not run on hardware
//
#include <hip/hip_runtime.h>
#include <hip/hip_bf16.h>

// ---------------------------------------------------------------------------
// CDNA5 (gfx1250) fused low-rank interaction + MLP
//   u = var_1 @ x1^T  (per row), v = x2 @ var_2        [B,32] each
//   x3[b, c*32+r] = v[b,c]*u[b,r]  (never materialized)
//   h1 = relu(x3@W1+b1); h2 = relu(h1@W2+b2); h3 = relu(h2@W3+b3)
//   out = h3@W4 + b4
// All GEMMs via v_wmma_f32_16x16x32_bf16 (wave32). One HBM pass over x1/x2.
// ---------------------------------------------------------------------------

typedef __attribute__((ext_vector_type(16))) __bf16 v16bf;
typedef __attribute__((ext_vector_type(8)))  __bf16 v8bf;
typedef __attribute__((ext_vector_type(8)))  float  v8f;
typedef __attribute__((ext_vector_type(4)))  float  vf4;

#define B_ROWS   32768
#define D_DIM    1024
#define WAVES    2            // waves per block; each wave owns 16 batch rows
#define TB       (WAVES * 16) // batch rows per block

__device__ __forceinline__ v8f wmma_bf16(v16bf a, v16bf b, v8f c) {
    // 8 args: (neg_a, A, neg_b, B, c_mod, C, reuse_a, reuse_b)
    return __builtin_amdgcn_wmma_f32_16x16x32_bf16(false, a, false, b,
                                                   (short)0, c, false, false);
}

// A fragment (16x32 bf16, M = lane&15) from an f32 row in global memory.
// ISA layout: lanes 0-15 elems 0..7 -> K=k0..k0+7, elems 8..15 -> K=k0+16..k0+23
//             lanes 16-31 shift K by +8 within each half.
__device__ __forceinline__ v16bf load_a_f32g(const float* row, int k0, int half) {
    vf4 q0 = *(const vf4*)(row + k0 + half * 8);
    vf4 q1 = *(const vf4*)(row + k0 + half * 8 + 4);
    vf4 q2 = *(const vf4*)(row + k0 + 16 + half * 8);
    vf4 q3 = *(const vf4*)(row + k0 + 16 + half * 8 + 4);
    v16bf a;
#pragma unroll
    for (int e = 0; e < 4; ++e) {
        a[e]      = (__bf16)q0[e];
        a[e + 4]  = (__bf16)q1[e];
        a[e + 8]  = (__bf16)q2[e];
        a[e + 12] = (__bf16)q3[e];
    }
    return a;
}

// A fragment from a bf16 row in LDS (row stride passed in by caller via ptr).
__device__ __forceinline__ v16bf load_a_lds(const __bf16* row, int k0, int half) {
    v8bf lo = *(const v8bf*)(row + k0 + half * 8);       // ds_load_b128
    v8bf hi = *(const v8bf*)(row + k0 + 16 + half * 8);  // ds_load_b128
    v16bf a;
#pragma unroll
    for (int e = 0; e < 8; ++e) { a[e] = lo[e]; a[e + 8] = hi[e]; }
    return a;
}

// B fragment (32x16 bf16, N = lane&15) from pre-transposed bf16 weights Wt[N][K]:
// lane holds K = k0+half*8..+7 and k0+16+half*8..+7 -> 2x b128 loads.
__device__ __forceinline__ v16bf load_bg(const __bf16* Wt, int K, int col,
                                         int k0, int half) {
    const __bf16* p = Wt + (size_t)col * K + k0 + half * 8;
    v8bf lo = *(const v8bf*)p;
    v8bf hi = *(const v8bf*)(p + 16);
    v16bf b;
#pragma unroll
    for (int e = 0; e < 8; ++e) { b[e] = lo[e]; b[e + 8] = hi[e]; }
    return b;
}

// Synthesize the x3 A-fragment for K-chunk c: x3[m][c*32 + r] = v[m][c]*u[m][r].
__device__ __forceinline__ v16bf gen_x3_frag(const float* urow, float vv, int half) {
    vf4 u0 = *(const vf4*)(urow + half * 8);
    vf4 u1 = *(const vf4*)(urow + half * 8 + 4);
    vf4 u2 = *(const vf4*)(urow + 16 + half * 8);
    vf4 u3 = *(const vf4*)(urow + 16 + half * 8 + 4);
    v16bf a;
#pragma unroll
    for (int e = 0; e < 4; ++e) {
        a[e]      = (__bf16)(vv * u0[e]);
        a[e + 4]  = (__bf16)(vv * u1[e]);
        a[e + 8]  = (__bf16)(vv * u2[e]);
        a[e + 12] = (__bf16)(vv * u3[e]);
    }
    return a;
}

// ---------------- prep kernels: f32 -> bf16 (copy / transpose) --------------

__global__ void cvt_copy_bf16(const float* __restrict__ src,
                              __bf16* __restrict__ dst, int n) {
    int i = blockIdx.x * 256 + threadIdx.x;
    if (i < n) dst[i] = (__bf16)src[i];
}

// src is [K][N] row-major f32; dst is [N][K] row-major bf16.
__global__ void cvt_transpose_bf16(const float* __restrict__ src,
                                   __bf16* __restrict__ dst, int K, int N) {
    int i = blockIdx.x * 256 + threadIdx.x;
    if (i < K * N) {
        int k = i / N, n = i - k * N;
        dst[(size_t)n * K + k] = (__bf16)src[i];
    }
}

// ---------------------------- fused main kernel -----------------------------

__global__ __launch_bounds__(WAVES * 32)
void fused_interaction_mlp(const float* __restrict__ x1,
                           const float* __restrict__ x2,
                           const __bf16* __restrict__ var1b,  // [32][1024]
                           const __bf16* __restrict__ var2t,  // [32][1024]
                           const __bf16* __restrict__ Wt1,    // [512][1024]
                           const __bf16* __restrict__ Wt2,    // [128][512]
                           const __bf16* __restrict__ Wt3,    // [32][128]
                           const float* __restrict__ b1,
                           const float* __restrict__ b2,
                           const float* __restrict__ b3,
                           const float* __restrict__ W4,      // [32][1] f32
                           const float* __restrict__ b4,
                           float* __restrict__ out) {
    __shared__ float  uvs[WAVES][16][64];   // cols 0..31 = u, 32..63 = v
    __shared__ __bf16 h1s[WAVES][16][512];
    __shared__ __bf16 h2s[WAVES][16][128];
    __shared__ float  h3s[WAVES][16][32];

    const int tid  = threadIdx.x;
    const int w    = tid >> 5;
    const int lane = tid & 31;
    const int n16  = lane & 15;   // A-row (M) and B/D-column (N) index
    const int half = lane >> 4;
    const int rowbase = blockIdx.x * TB + w * 16;

    const float* x1row = x1 + (size_t)(rowbase + n16) * D_DIM;
    const float* x2row = x2 + (size_t)(rowbase + n16) * D_DIM;

    // ---- Phase 1: u = x1 @ var_1^T, v = x2 @ var_2 (f32 acc, bf16 WMMA) ----
    v8f au0 = {}, au1 = {}, av0 = {}, av1 = {};
#pragma unroll 1
    for (int kc = 0; kc < D_DIM; kc += 32) {
        v16bf a1  = load_a_f32g(x1row, kc, half);
        v16bf a2  = load_a_f32g(x2row, kc, half);
        v16bf bu0 = load_bg(var1b, D_DIM, n16,      kc, half);
        v16bf bu1 = load_bg(var1b, D_DIM, n16 + 16, kc, half);
        v16bf bv0 = load_bg(var2t, D_DIM, n16,      kc, half);
        v16bf bv1 = load_bg(var2t, D_DIM, n16 + 16, kc, half);
        au0 = wmma_bf16(a1, bu0, au0);
        au1 = wmma_bf16(a1, bu1, au1);
        av0 = wmma_bf16(a2, bv0, av0);
        av1 = wmma_bf16(a2, bv1, av1);
    }
    // D layout: VGPR r -> M = r + 8*half, N = lane&15
#pragma unroll
    for (int r = 0; r < 8; ++r) {
        const int m = r + 8 * half;
        uvs[w][m][n16]      = au0[r];
        uvs[w][m][16 + n16] = au1[r];
        uvs[w][m][32 + n16] = av0[r];
        uvs[w][m][48 + n16] = av1[r];
    }
    __syncthreads();

    // ---- Phase 2: h1 = relu(x3 @ W1 + b1), x3 synthesized from u,v ----
    const float* urow = &uvs[w][n16][0];
    const float* vrow = &uvs[w][n16][32];
#pragma unroll 1
    for (int nt = 0; nt < 32; ++nt) {
        const int col = nt * 16 + n16;
        const float bias = b1[col];
        v8f acc;
#pragma unroll
        for (int r = 0; r < 8; ++r) acc[r] = bias;
#pragma unroll 1
        for (int c = 0; c < 32; ++c) {
            v16bf a = gen_x3_frag(urow, vrow[c], half);
            v16bf b = load_bg(Wt1, 1024, col, c * 32, half);
            acc = wmma_bf16(a, b, acc);
        }
#pragma unroll
        for (int r = 0; r < 8; ++r) {
            float val = acc[r];
            h1s[w][r + 8 * half][col] = (__bf16)(val > 0.f ? val : 0.f);
        }
    }
    __syncthreads();

    // ---- Phase 3: h2 = relu(h1 @ W2 + b2) ----
    const __bf16* h1row = &h1s[w][n16][0];
#pragma unroll 1
    for (int nt = 0; nt < 8; ++nt) {
        const int col = nt * 16 + n16;
        const float bias = b2[col];
        v8f acc;
#pragma unroll
        for (int r = 0; r < 8; ++r) acc[r] = bias;
#pragma unroll 1
        for (int kc = 0; kc < 512; kc += 32) {
            v16bf a = load_a_lds(h1row, kc, half);
            v16bf b = load_bg(Wt2, 512, col, kc, half);
            acc = wmma_bf16(a, b, acc);
        }
#pragma unroll
        for (int r = 0; r < 8; ++r) {
            float val = acc[r];
            h2s[w][r + 8 * half][col] = (__bf16)(val > 0.f ? val : 0.f);
        }
    }
    __syncthreads();

    // ---- Phase 4: h3 = relu(h2 @ W3 + b3) ----
    const __bf16* h2row = &h2s[w][n16][0];
#pragma unroll 1
    for (int nt = 0; nt < 2; ++nt) {
        const int col = nt * 16 + n16;
        const float bias = b3[col];
        v8f acc;
#pragma unroll
        for (int r = 0; r < 8; ++r) acc[r] = bias;
#pragma unroll
        for (int kc = 0; kc < 128; kc += 32) {
            v16bf a = load_a_lds(h2row, kc, half);
            v16bf b = load_bg(Wt3, 128, col, kc, half);
            acc = wmma_bf16(a, b, acc);
        }
#pragma unroll
        for (int r = 0; r < 8; ++r) {
            float val = acc[r];
            h3s[w][r + 8 * half][col] = val > 0.f ? val : 0.f;
        }
    }
    __syncthreads();

    // ---- Phase 5: out = h3 @ W4 + b4 (trivial 32->1 dot per row) ----
    if (lane < 16) {
        float s = b4[0];
#pragma unroll
        for (int r = 0; r < 32; ++r) s += h3s[w][lane][r] * W4[r];
        out[rowbase + lane] = s;
    }
}

// ---------------------------------------------------------------------------

extern "C" void kernel_launch(void* const* d_in, const int* in_sizes, int n_in,
                              void* d_out, int out_size, void* d_ws, size_t ws_size,
                              hipStream_t stream) {
    const float* x1    = (const float*)d_in[0];
    const float* x2    = (const float*)d_in[1];
    const float* var_1 = (const float*)d_in[2];   // [32,1024]
    const float* var_2 = (const float*)d_in[3];   // [1024,32]
    const float* W1    = (const float*)d_in[4];   // [1024,512]
    const float* b1    = (const float*)d_in[5];
    const float* W2    = (const float*)d_in[6];   // [512,128]
    const float* b2    = (const float*)d_in[7];
    const float* W3    = (const float*)d_in[8];   // [128,32]
    const float* b3    = (const float*)d_in[9];
    const float* W4    = (const float*)d_in[10];  // [32,1]
    const float* b4    = (const float*)d_in[11];
    float* out = (float*)d_out;

    // Workspace layout (bf16, ~1.26 MB total)
    __bf16* ws    = (__bf16*)d_ws;
    __bf16* var1b = ws;                    // 32*1024
    __bf16* var2t = var1b + 32 * 1024;     // 32*1024  (var_2 transposed)
    __bf16* Wt1   = var2t + 32 * 1024;     // 512*1024 (W1 transposed)
    __bf16* Wt2   = Wt1 + 512 * 1024;      // 128*512
    __bf16* Wt3   = Wt2 + 128 * 512;       // 32*128

    cvt_copy_bf16<<<(32 * 1024 + 255) / 256, 256, 0, stream>>>(var_1, var1b, 32 * 1024);
    cvt_transpose_bf16<<<(32 * 1024 + 255) / 256, 256, 0, stream>>>(var_2, var2t, 1024, 32);
    cvt_transpose_bf16<<<(512 * 1024 + 255) / 256, 256, 0, stream>>>(W1, Wt1, 1024, 512);
    cvt_transpose_bf16<<<(128 * 512 + 255) / 256, 256, 0, stream>>>(W2, Wt2, 512, 128);
    cvt_transpose_bf16<<<(32 * 128 + 255) / 256, 256, 0, stream>>>(W3, Wt3, 128, 32);

    fused_interaction_mlp<<<B_ROWS / TB, WAVES * 32, 0, stream>>>(
        x1, x2, var1b, var2t, Wt1, Wt2, Wt3, b1, b2, b3, W4, b4, out);
}
